// Attribute_Decoder_6141803233544
// MI455X (gfx1250) — matrix operand order repaired and checked by reading, not compiled, not run on hardware
//
#include <hip/hip_runtime.h>
#include <hip/hip_bf16.h>

typedef __attribute__((ext_vector_type(16))) _Float16 v16h;
typedef __attribute__((ext_vector_type(8)))  float    v8f;

#define GCN_N 50000

// ---------------------------------------------------------------------------
// Pack W (f32, row-major KxN) into WMMA-B-fragment-ready f16 layout:
//   Wp[(kstep*N + n)*32 + j] = (f16) W[(kstep*32 + j)*N + n]
// Lane (col n, half 'hi') then reads its 16 K-values as one 32B-aligned run.
// ---------------------------------------------------------------------------
__global__ void pack_w_kernel(const float* __restrict__ W, _Float16* __restrict__ Wp,
                              int N, int K) {
  int idx = blockIdx.x * blockDim.x + threadIdx.x;
  int total = K * N;
  if (idx >= total) return;
  int j     = idx & 31;
  int rem   = idx >> 5;        // kstep*N + n
  int n     = rem % N;
  int kstep = rem / N;
  Wp[idx] = (_Float16)W[(kstep * 32 + j) * N + n];
}

// ---------------------------------------------------------------------------
// WMMA GEMM: C[M x N] = A[M x K] (f32 -> f16) * Wp (packed f16), f32 accum.
// One wave computes 16 rows x 128 cols (8 16x16x32 tiles).
// N, K compile-time: fully unrolled K loop, all addresses immediate offsets.
// ---------------------------------------------------------------------------
template <int N, int K>
__global__ void gemm_wmma_kernel(const float* __restrict__ A,
                                 const _Float16* __restrict__ Bp,
                                 float* __restrict__ C, int M) {
  const int lane   = threadIdx.x & 31;
  const int wave   = threadIdx.x >> 5;
  const int wpb    = blockDim.x >> 5;
  const int rowBlk = blockIdx.x * wpb + wave;
  const int nRowBlk = M >> 4;
  if (rowBlk >= nRowBlk) return;               // wave-uniform guard (EXEC stays full)

  const unsigned colBase = blockIdx.y * 128u;
  const unsigned l15 = lane & 15;
  const unsigned hi  = lane >> 4;              // 0 or 1
  const unsigned m   = ((unsigned)rowBlk << 4) + l15;

  // per-lane base pointers; everything below is immediate offsets off these
  const float*    a0 = A  + (size_t)m * K + (hi << 4);              // +16 floats if hi
  const v16h*     bb = (const v16h*)(Bp + (((size_t)(colBase + l15)) << 5) + (hi << 4));
  float*          cb = C  + (size_t)(((unsigned)rowBlk << 4) + (hi << 3)) * N + colBase + l15;

  v8f acc[8];
#pragma unroll
  for (int t = 0; t < 8; ++t) acc[t] = (v8f)0.0f;

#pragma unroll
  for (int kk = 0; kk < K / 32; ++kk) {
    // ---- A fragment: row m, K = 32*kk + 16*hi + {0..15}
    const float4* a4 = (const float4*)(a0 + (kk << 5));
    float4 p0 = a4[0], p1 = a4[1], p2 = a4[2], p3 = a4[3];
    v16h a;
    a[0]  = (_Float16)p0.x; a[1]  = (_Float16)p0.y; a[2]  = (_Float16)p0.z; a[3]  = (_Float16)p0.w;
    a[4]  = (_Float16)p1.x; a[5]  = (_Float16)p1.y; a[6]  = (_Float16)p1.z; a[7]  = (_Float16)p1.w;
    a[8]  = (_Float16)p2.x; a[9]  = (_Float16)p2.y; a[10] = (_Float16)p2.z; a[11] = (_Float16)p2.w;
    a[12] = (_Float16)p3.x; a[13] = (_Float16)p3.y; a[14] = (_Float16)p3.z; a[15] = (_Float16)p3.w;

    // ---- preload ALL 8 B fragments of this k-step (distinct registers,
    //      one load clause, one wait; tile stride = 512 halfs = 32 v16h units... )
    //      v16h unit = 32B; per-tile stride = 1024B = 32 units; per-kk = 64*N bytes.
    v16h b[8];
#pragma unroll
    for (int t = 0; t < 8; ++t)
      b[t] = bb[(size_t)kk * (N * 2) + t * 32];   // (64*N bytes)/32 = 2N units

#pragma unroll
    for (int t = 0; t < 8; ++t)
      acc[t] = __builtin_amdgcn_wmma_f32_16x16x32_f16(
          /*neg_a=*/false, a, /*neg_b=*/false, b[t],
          /*c_mod=*/(short)0, acc[t], /*reuse_a=*/false, /*reuse_b=*/false);
  }

  // ---- store: acc[t][r] -> C[rowBlk*16 + r + 8*hi][colBase + t*16 + l15]
#pragma unroll
  for (int t = 0; t < 8; ++t)
#pragma unroll
    for (int r = 0; r < 8; ++r)
      cb[(size_t)r * N + t * 16] = acc[t][r];
}

// ---------------------------------------------------------------------------
// Degree / normalization
// ---------------------------------------------------------------------------
__global__ void init_deg_kernel(float* deg, int n) {
  int i = blockIdx.x * blockDim.x + threadIdx.x;
  if (i < n) deg[i] = 1.0f;                     // self-loop
}
__global__ void deg_accum_kernel(const int* __restrict__ dst, float* deg, int E) {
  int e = blockIdx.x * blockDim.x + threadIdx.x;
  if (e < E) atomicAdd(&deg[dst[e]], 1.0f);
}
__global__ void dinv_kernel(const float* __restrict__ deg, float* __restrict__ dinv, int n) {
  int i = blockIdx.x * blockDim.x + threadIdx.x;
  if (i < n) dinv[i] = rsqrtf(deg[i]);          // deg >= 1 always
}

// Out[i][:] = dinv[i]^2 * H[i][:]   (self-loop message, also initializes Out)
__global__ void selfloop_init_kernel(const float* __restrict__ H, const float* __restrict__ dinv,
                                     float* __restrict__ Out, int n, int Cq) {
  int idx = blockIdx.x * blockDim.x + threadIdx.x;
  if (idx >= n * Cq) return;
  int node = idx / Cq;
  float w = dinv[node]; w *= w;
  float4 v = ((const float4*)H)[idx];
  float4 o; o.x = v.x * w; o.y = v.y * w; o.z = v.z * w; o.w = v.w * w;
  ((float4*)Out)[idx] = o;
}

// One wave per edge: coalesced float4 gather of H[src], scaled atomic adds into Out[dst].
__global__ void scatter_edges_kernel(const float* __restrict__ H, const int* __restrict__ src,
                                     const int* __restrict__ dst, const float* __restrict__ dinv,
                                     float* __restrict__ Out, int E, int Cq) {
  int wid  = (blockIdx.x * blockDim.x + threadIdx.x) >> 5;
  int lane = threadIdx.x & 31;
  if (wid >= E) return;                         // wave-uniform
  int s = src[wid];
  int d = dst[wid];
  float nrm = dinv[s] * dinv[d];
  const float4* hs = (const float4*)(H + (size_t)s * (Cq * 4));
  float* od = Out + (size_t)d * (Cq * 4);
  for (int q = lane; q < Cq; q += 32) {
    float4 v = hs[q];
    atomicAdd(od + q * 4 + 0, v.x * nrm);
    atomicAdd(od + q * 4 + 1, v.y * nrm);
    atomicAdd(od + q * 4 + 2, v.z * nrm);
    atomicAdd(od + q * 4 + 3, v.w * nrm);
  }
}

__global__ void bias_relu_kernel(float* __restrict__ X, const float* __restrict__ b,
                                 int total, int Cc) {
  int idx = blockIdx.x * blockDim.x + threadIdx.x;
  if (idx >= total) return;
  float v = X[idx] + b[idx % Cc];
  X[idx] = v > 0.0f ? v : 0.0f;
}

// ---------------------------------------------------------------------------
extern "C" void kernel_launch(void* const* d_in, const int* in_sizes, int n_in,
                              void* d_out, int out_size, void* d_ws, size_t ws_size,
                              hipStream_t stream) {
  const float* x    = (const float*)d_in[0];   // [N,128]
  const int*   eidx = (const int*)d_in[1];     // [2,E]
  const float* W1   = (const float*)d_in[2];   // [128,128]
  const float* b1   = (const float*)d_in[3];   // [128]
  const float* W2   = (const float*)d_in[4];   // [128,256]
  const float* b2   = (const float*)d_in[5];   // [256]
  float* out = (float*)d_out;                  // [N,256]

  const int N  = GCN_N;
  const int E  = in_sizes[1] / 2;
  const int C1 = 128, C2 = 256, K = 128;
  const int* esrc = eidx;
  const int* edst = eidx + E;

  // ---- workspace carve (all 32B aligned by construction)
  char* ws = (char*)d_ws;
  float* deg  = (float*)ws; ws += (size_t)N * 4;
  float* dinv = (float*)ws; ws += (size_t)N * 4;
  float* H1   = (float*)ws; ws += (size_t)N * C1 * 4;
  float* A1   = (float*)ws; ws += (size_t)N * C1 * 4;   // scatter acc, then ReLU'd hidden
  float* H2   = (float*)ws; ws += (size_t)N * C2 * 4;
  _Float16* Wp1 = (_Float16*)ws; ws += (size_t)K * C1 * 2;
  _Float16* Wp2 = (_Float16*)ws; ws += (size_t)K * C2 * 2;

  const int TB = 256;
  // ---- pack weights into WMMA-B layout
  pack_w_kernel<<<(K * C1 + TB - 1) / TB, TB, 0, stream>>>(W1, Wp1, C1, K);
  pack_w_kernel<<<(K * C2 + TB - 1) / TB, TB, 0, stream>>>(W2, Wp2, C2, K);

  // ---- degrees & normalization (graph is identical for both layers)
  init_deg_kernel<<<(N + TB - 1) / TB, TB, 0, stream>>>(deg, N);
  deg_accum_kernel<<<(E + TB - 1) / TB, TB, 0, stream>>>(edst, deg, E);
  dinv_kernel<<<(N + TB - 1) / TB, TB, 0, stream>>>(deg, dinv, N);

  const int nRowBlk = N / 16;                    // 3125 (exact)
  const int gemmBlocks = (nRowBlk + 3) / 4;      // 4 waves / block of 128

  // ================= layer 1 =================
  gemm_wmma_kernel<128, 128><<<dim3(gemmBlocks, 1), 128, 0, stream>>>(x, Wp1, H1, N);
  {
    int Cq = C1 / 4;
    selfloop_init_kernel<<<(N * Cq + TB - 1) / TB, TB, 0, stream>>>(H1, dinv, A1, N, Cq);
    scatter_edges_kernel<<<(E * 32 + TB - 1) / TB, TB, 0, stream>>>(H1, esrc, edst, dinv, A1, E, Cq);
    bias_relu_kernel<<<(N * C1 + TB - 1) / TB, TB, 0, stream>>>(A1, b1, N * C1, C1);
  }

  // ================= layer 2 =================
  gemm_wmma_kernel<256, 128><<<dim3(gemmBlocks, 2), 128, 0, stream>>>(A1, Wp2, H2, N);
  {
    int Cq = C2 / 4;
    selfloop_init_kernel<<<(N * Cq + TB - 1) / TB, TB, 0, stream>>>(H2, dinv, out, N, Cq);
    scatter_edges_kernel<<<(E * 32 + TB - 1) / TB, TB, 0, stream>>>(H2, esrc, edst, dinv, out, E, Cq);
    bias_relu_kernel<<<(N * C2 + TB - 1) / TB, TB, 0, stream>>>(out, b2, N * C2, C2);
  }
}
